// GCN_49718541418973
// MI455X (gfx1250) — matrix operand order repaired and checked by reading, not compiled, not run on hardware
//
#include <hip/hip_runtime.h>
#include <hip/hip_bf16.h>

#define NN   50000
#define EE   800000
#define HIDD 128
#define GG   256

typedef __attribute__((ext_vector_type(16))) __bf16 v16bf;
typedef __attribute__((ext_vector_type(8)))  __bf16 bf16x8;
typedef __attribute__((ext_vector_type(8)))  float  v8f;

// ---------------------------------------------------------------- utilities
__global__ void fill_f32(float* __restrict__ p, long n, float v) {
    long i = (long)blockIdx.x * blockDim.x + threadIdx.x;
    if (i < n) p[i] = v;
}

__global__ void deg_accum(const int* __restrict__ ei, float* __restrict__ deg) {
    int e = blockIdx.x * blockDim.x + threadIdx.x;
    if (e < EE) atomicAdd(&deg[ei[EE + e]], 1.0f);   // dst row of edge_index
}

__global__ void dinv_k(const float* __restrict__ deg, float* __restrict__ dinv) {
    int i = blockIdx.x * blockDim.x + threadIdx.x;
    if (i < NN) dinv[i] = rsqrtf(deg[i]);            // deg >= 1 (self-loop)
}

// ------------------------------------------------- dense X@W via WMMA bf16
// A: [NN,128] f32 activations, W: [128,128] f32 (k-major), out: [NN,128] f32
__global__ __launch_bounds__(128) void gcn_gemm_wmma(const float* __restrict__ A,
                                                     const float* __restrict__ W,
                                                     float* __restrict__ out) {
    __shared__ __bf16 wt[HIDD * HIDD];       // transposed weights [n][k], 32KB
    __shared__ __bf16 at[4][16 * HIDD];      // per-wave A tiles, 16KB

    const int tid  = threadIdx.x;
    const int wave = tid >> 5;
    const int lane = tid & 31;

    // stage W transposed as bf16 (coalesced global reads)
    for (int i = tid; i < HIDD * HIDD; i += 128) {
        int k = i >> 7, n = i & 127;
        wt[n * HIDD + k] = (__bf16)W[i];
    }
    const int rt = blockIdx.x * 4 + wave;    // 16-row tile index
    if (rt * 16 < NN) {
        const float* arow = A + (size_t)rt * 16 * HIDD;
        for (int i = lane; i < 16 * HIDD; i += 32)
            at[wave][i] = (__bf16)arow[i];
    }
    __syncthreads();
    if (rt * 16 >= NN) return;

    // A fragments per CDNA5 ISA 16-bit A 16x32 layout
    const int row  = lane & 15;
    const int koff = (lane >> 4) << 3;       // lanes 16-31 hold K=8..15 / 24..31
    v16bf afrag[4];
#pragma unroll
    for (int kc = 0; kc < 4; ++kc) {
        bf16x8 lo = *(const bf16x8*)&at[wave][row * HIDD + kc * 32 + koff];
        bf16x8 hi = *(const bf16x8*)&at[wave][row * HIDD + kc * 32 + 16 + koff];
#pragma unroll
        for (int i = 0; i < 8; ++i) { afrag[kc][i] = lo[i]; afrag[kc][i + 8] = hi[i]; }
    }

    const int bcol = lane & 15;
    const int bk   = (lane >> 4) << 3;       // lanes 16-31 hold K=8..15 / 24..31
    const int rbase = rt * 16 + ((lane >> 4) << 3);
#pragma unroll
    for (int nt = 0; nt < 8; ++nt) {
        v8f acc = {};
#pragma unroll
        for (int kc = 0; kc < 4; ++kc) {
            const __bf16* bp = &wt[(nt * 16 + bcol) * HIDD + kc * 32 + bk];
            bf16x8 lo = *(const bf16x8*)bp;
            bf16x8 hi = *(const bf16x8*)(bp + 16);
            v16bf bfrag;
#pragma unroll
            for (int i = 0; i < 8; ++i) { bfrag[i] = lo[i]; bfrag[i + 8] = hi[i]; }
            acc = __builtin_amdgcn_wmma_f32_16x16x32_bf16(
                false, afrag[kc], false, bfrag, (short)0, acc, false, false);
        }
#pragma unroll
        for (int r = 0; r < 8; ++r)
            out[(size_t)(rbase + r) * HIDD + nt * 16 + bcol] = acc[r];
    }
}

// self-loop term initializes agg (pure write, replaces zero-fill + later RMW);
// runs BEFORE edge_scatter — stream order makes this safe.
__global__ void self_loop_init(const float* __restrict__ xw,
                               const float* __restrict__ dinv,
                               float* __restrict__ agg) {
    long i = (long)blockIdx.x * blockDim.x + threadIdx.x;
    if (i >= (long)NN * HIDD) return;
    float dv = dinv[i >> 7];
    agg[i] = xw[i] * dv * dv;
}

// --------------------------------------------- sparse scatter: wave per edge
__global__ void edge_scatter(const int* __restrict__ ei,
                             const float* __restrict__ xw,
                             const float* __restrict__ dinv,
                             float* __restrict__ agg) {
    int gid  = blockIdx.x * blockDim.x + threadIdx.x;
    int e    = gid >> 5;
    int lane = gid & 31;
    if (e >= EE) return;
    int s = ei[e], d = ei[EE + e];
    float nrm = dinv[s] * dinv[d];
    const float4 v = *(const float4*)(xw + (size_t)s * HIDD + lane * 4);
    float* base = agg + (size_t)d * HIDD + lane * 4;
    atomicAdd(base + 0, v.x * nrm);
    atomicAdd(base + 1, v.y * nrm);
    atomicAdd(base + 2, v.z * nrm);
    atomicAdd(base + 3, v.w * nrm);
}

__global__ void bias_relu(const float* __restrict__ agg,
                          const float* __restrict__ b,
                          float* __restrict__ h) {
    long i = (long)blockIdx.x * blockDim.x + threadIdx.x;
    if (i >= (long)NN * HIDD) return;
    h[i] = fmaxf(agg[i] + b[i & 127], 0.0f);
}

// ----------------------------------------------------- pooling: wave per node
__global__ void pool_k(const float* __restrict__ h, const int* __restrict__ batch,
                       float* __restrict__ psum, float* __restrict__ pmax,
                       float* __restrict__ cnt) {
    int gid  = blockIdx.x * blockDim.x + threadIdx.x;
    int node = gid >> 5;
    int lane = gid & 31;
    if (node >= NN) return;
    int g = batch[node];
    const float4 v = *(const float4*)(h + (size_t)node * HIDD + lane * 4);
    float* sb = psum + (size_t)g * HIDD + lane * 4;
    unsigned* mb = (unsigned*)(pmax + (size_t)g * HIDD + lane * 4);
    atomicAdd(sb + 0, v.x); atomicAdd(sb + 1, v.y);
    atomicAdd(sb + 2, v.z); atomicAdd(sb + 3, v.w);
    // post-ReLU values are >= 0, so IEEE order == unsigned-int order; init 0
    atomicMax(mb + 0, __float_as_uint(v.x));
    atomicMax(mb + 1, __float_as_uint(v.y));
    atomicMax(mb + 2, __float_as_uint(v.z));
    atomicMax(mb + 3, __float_as_uint(v.w));
    if (lane == 0) atomicAdd(&cnt[g], 1.0f);
}

// -------------------------------------------- per-graph MLP + log_softmax
__global__ __launch_bounds__(128) void mlp_k(const float* __restrict__ psum,
                                             const float* __restrict__ pmax,
                                             const float* __restrict__ cnt,
                                             const float* __restrict__ l1w,
                                             const float* __restrict__ l1b,
                                             const float* __restrict__ l2w,
                                             const float* __restrict__ l2b,
                                             float* __restrict__ out) {
    __shared__ float gf[3 * HIDD];
    __shared__ float hid[HIDD];
    const int g = blockIdx.x, j = threadIdx.x;
    float c = fmaxf(cnt[g], 1.0f);
    float s = psum[(size_t)g * HIDD + j];
    gf[j]            = s;
    gf[HIDD + j]     = s / c;
    gf[2 * HIDD + j] = pmax[(size_t)g * HIDD + j];
    __syncthreads();
    float acc = l1b[j];
    for (int k = 0; k < 3 * HIDD; ++k) acc += gf[k] * l1w[k * HIDD + j];
    hid[j] = fmaxf(acc, 0.0f);
    __syncthreads();
    if (j == 0) {
        float l0 = l2b[0], l1 = l2b[1];
        for (int k = 0; k < HIDD; ++k) {
            l0 += hid[k] * l2w[k * 2 + 0];
            l1 += hid[k] * l2w[k * 2 + 1];
        }
        float m  = fmaxf(l0, l1);
        float ls = m + logf(expf(l0 - m) + expf(l1 - m));
        out[g * 2 + 0] = l0 - ls;
        out[g * 2 + 1] = l1 - ls;
    }
}

// ----------------------------------------------------------------- launcher
extern "C" void kernel_launch(void* const* d_in, const int* in_sizes, int n_in,
                              void* d_out, int out_size, void* d_ws, size_t ws_size,
                              hipStream_t stream) {
    const float* x    = (const float*)d_in[0];
    const int*   ei   = (const int*)d_in[1];     // [2,E] flattened: src row, dst row
    const int*   bat  = (const int*)d_in[2];
    const float* W0   = (const float*)d_in[3];
    const float* b0   = (const float*)d_in[4];
    const float* Wh   = (const float*)d_in[5];   // [3,128,128]
    const float* bh   = (const float*)d_in[6];   // [3,128]
    const float* l1w  = (const float*)d_in[7];
    const float* l1b  = (const float*)d_in[8];
    const float* l2w  = (const float*)d_in[9];
    const float* l2b  = (const float*)d_in[10];
    float* out = (float*)d_out;

    float* ws   = (float*)d_ws;
    float* deg  = ws;  ws += NN;
    float* dinv = ws;  ws += NN;
    float* h    = ws;  ws += (size_t)NN * HIDD;
    float* xw   = ws;  ws += (size_t)NN * HIDD;
    float* agg  = ws;  ws += (size_t)NN * HIDD;
    float* psum = ws;  ws += (size_t)GG * HIDD;
    float* pmax = ws;  ws += (size_t)GG * HIDD;
    float* cnt  = ws;  ws += GG;

    const long NH = (long)NN * HIDD;
    const int  nh_blocks = (int)((NH + 255) / 256);

    // degree (init 1.0 accounts for self-loop) and D^-1/2
    fill_f32<<<(NN + 255) / 256, 256, 0, stream>>>(deg, NN, 1.0f);
    deg_accum<<<EE / 256, 256, 0, stream>>>(ei, deg);
    dinv_k<<<(NN + 255) / 256, 256, 0, stream>>>(deg, dinv);

    const int gemm_blocks = (NN / 16 + 3) / 4;          // 4 waves, 1 row-tile each
    for (int l = 0; l < 4; ++l) {
        const float* in = (l == 0) ? x : h;
        const float* W  = (l == 0) ? W0 : Wh + (size_t)(l - 1) * HIDD * HIDD;
        const float* b  = (l == 0) ? b0 : bh + (size_t)(l - 1) * HIDD;
        gcn_gemm_wmma<<<gemm_blocks, 128, 0, stream>>>(in, W, xw);
        self_loop_init<<<nh_blocks, 256, 0, stream>>>(xw, dinv, agg);   // agg = xw*dinv^2
        edge_scatter<<<(EE * 32) / 256, 256, 0, stream>>>(ei, xw, dinv, agg);
        bias_relu<<<nh_blocks, 256, 0, stream>>>(agg, b, h);
    }

    // pooling buffers are contiguous: psum | pmax | cnt
    fill_f32<<<(2 * GG * HIDD + GG + 255) / 256, 256, 0, stream>>>(psum,
                                                                   2 * GG * HIDD + GG, 0.0f);
    pool_k<<<(NN * 32) / 256, 256, 0, stream>>>(h, bat, psum, pmax, cnt);
    mlp_k<<<GG, 128, 0, stream>>>(psum, pmax, cnt, l1w, l1b, l2w, l2b, out);
}
